// G_Block_39350490365958
// MI455X (gfx1250) — matrix operand order repaired and checked by reading, not compiled, not run on hardware
//
#include <hip/hip_runtime.h>
#include <hip/hip_bf16.h>
#include <math.h>

typedef __attribute__((ext_vector_type(16))) __bf16 v16bf;
typedef __attribute__((ext_vector_type(8)))  __bf16 v8bf;
typedef __attribute__((ext_vector_type(8)))  float  v8f;

#define NB   8
#define CH   512      // CIN = CINT = COUT = 512
#define HIN  32
#define HOUT 64
#define PD   66       // padded/stuffed spatial extent
#define TAPS 9

// LDS tile: 3 rows x 66 x-positions x 32 channels of bf16, double buffered
#define TILE_ELEMS (3 * PD * 32)            // 6336 bf16 = 12672 B
#define TILE_B128  (TILE_ELEMS / 8)         // 792 16-byte transfers

__device__ __forceinline__ float lrelu(float v) { return v > 0.f ? v : 0.2f * v; }

#define WMMA_BF16(a, b, c) \
    __builtin_amdgcn_wmma_f32_16x16x32_bf16(false, (a), false, (b), (short)0, (c), false, false)

// ---------------------------------------------------------------------------
// s = v @ sw + sb  for the three modulation vectors (up, conv, rgb)
// ---------------------------------------------------------------------------
__global__ void modstyle_kernel(const float* __restrict__ v,
                                const float* __restrict__ up_sw, const float* __restrict__ up_sb,
                                const float* __restrict__ c_sw,  const float* __restrict__ c_sb,
                                const float* __restrict__ rgb_sw,const float* __restrict__ rgb_sb,
                                float* __restrict__ s_up, float* __restrict__ s_c,
                                float* __restrict__ s_rgb)
{
    int t = blockIdx.x * blockDim.x + threadIdx.x;
    if (t >= 3 * NB * CH) return;
    int c = t & (CH - 1);
    int n = (t >> 9) & (NB - 1);
    int which = t >> 12;
    const float* sw = (which == 0) ? up_sw : (which == 1) ? c_sw : rgb_sw;
    const float* sb = (which == 0) ? up_sb : (which == 1) ? c_sb : rgb_sb;
    float acc = sb[c];
    const float* vp = v + n * CH;
    for (int l = 0; l < CH; ++l) acc += vp[l] * sw[l * CH + c];
    float* dst = (which == 0) ? s_up : (which == 1) ? s_c : s_rgb;
    dst[n * CH + c] = acc;
}

// ---------------------------------------------------------------------------
// Wsq[co][ci] = sum_hw W^2  (up_w is [in][out][3][3], c_w is [out][in][3][3])
// ---------------------------------------------------------------------------
__global__ void wsq_kernel(const float* __restrict__ up_w, const float* __restrict__ c_w,
                           float* __restrict__ wsq_up, float* __restrict__ wsq_c)
{
    int t = blockIdx.x * blockDim.x + threadIdx.x;
    if (t >= 2 * CH * CH) return;
    int ci = t & (CH - 1);
    int co = (t >> 9) & (CH - 1);
    int which = t >> 18;
    float a = 0.f;
    if (which == 0) {
        const float* p = up_w + ((size_t)ci * CH + co) * TAPS;
        for (int k = 0; k < TAPS; ++k) a += p[k] * p[k];
        wsq_up[co * CH + ci] = a;
    } else {
        const float* p = c_w + ((size_t)co * CH + ci) * TAPS;
        for (int k = 0; k < TAPS; ++k) a += p[k] * p[k];
        wsq_c[co * CH + ci] = a;
    }
}

// ---------------------------------------------------------------------------
// isg[n][co] = 1/sqrt( sum_ci s^2 * Wsq + 1e-8 )
// ---------------------------------------------------------------------------
__global__ void invsigma_kernel(const float* __restrict__ s_up, const float* __restrict__ s_c,
                                const float* __restrict__ wsq_up, const float* __restrict__ wsq_c,
                                float* __restrict__ isg_up, float* __restrict__ isg_c)
{
    int t = blockIdx.x * blockDim.x + threadIdx.x;
    if (t >= 2 * NB * CH) return;
    int co = t & (CH - 1);
    int n  = (t >> 9) & (NB - 1);
    int which = t >> 12;
    const float* s = which ? s_c : s_up;
    const float* w = which ? wsq_c : wsq_up;
    float a = 1e-8f;
    const float* sp = s + n * CH;
    const float* wp = w + (size_t)co * CH;
    for (int ci = 0; ci < CH; ++ci) { float sv = sp[ci]; a += sv * sv * wp[ci]; }
    (which ? isg_c : isg_up)[n * CH + co] = 1.0f / sqrtf(a);
}

// ---------------------------------------------------------------------------
// Pack weights as bf16 [co][tap][ci].  For the up-conv: Wt[o][i][ky][kx] =
// up_w[i][o][2-ky][2-kx] (flip + transpose of the stored [in][out][k][k]).
// ---------------------------------------------------------------------------
__global__ void wpack_kernel(const float* __restrict__ up_w, const float* __restrict__ c_w,
                             __bf16* __restrict__ wp_up, __bf16* __restrict__ wp_c)
{
    int t = blockIdx.x * blockDim.x + threadIdx.x;
    if (t >= 2 * CH * TAPS * CH) return;
    int ci = t & (CH - 1);
    int r = t >> 9;
    int tap = r % TAPS; r /= TAPS;
    int co = r & (CH - 1);
    int which = r >> 9;
    if (which == 0) {
        int ky = tap / 3, kx = tap % 3;
        float w = up_w[((size_t)ci * CH + co) * TAPS + (2 - ky) * 3 + (2 - kx)];
        wp_up[((size_t)co * TAPS + tap) * CH + ci] = (__bf16)w;
    } else {
        wp_c[((size_t)co * TAPS + tap) * CH + ci] =
            (__bf16)c_w[((size_t)co * CH + ci) * TAPS + tap];
    }
}

// ---------------------------------------------------------------------------
// Zero-stuff modulated x into padded NHWC bf16 buffer (odd ty,tx = x * s_up).
// Buffer was memset to zero beforehand.
// ---------------------------------------------------------------------------
__global__ void stuff_kernel(const float* __restrict__ x, const float* __restrict__ s_up,
                             __bf16* __restrict__ a1)
{
    int t = blockIdx.x * blockDim.x + threadIdx.x;
    if (t >= NB * HIN * HIN * CH) return;
    int c = t & (CH - 1);
    int r = t >> 9;
    int ix = r & 31; r >>= 5;
    int iy = r & 31;
    int n = r >> 5;
    float v = x[(((size_t)n * CH + c) * HIN + iy) * HIN + ix] * s_up[n * CH + c];
    a1[(((size_t)(n * PD + 2 * iy + 1)) * PD + 2 * ix + 1) * CH + c] = (__bf16)v;
}

// ---------------------------------------------------------------------------
// Implicit-GEMM 3x3 conv on WMMA bf16 (fp32 accumulate).
//   in : [NB][PD][PD][CH] bf16 (NHWC, halo included)
//   wp : [CH][9][CH] bf16
// Block: 256 threads = 8 waves; each wave computes 32 couts x 64 positions
// (8 accumulator tiles) -> 256 couts per block, one output row per block.
// B tile (3 x 66 x 32ch) staged in LDS via global_load_async_to_lds_b128,
// double-buffered across the 16 channel chunks.  A fragments are register
// double-buffered across taps.  Per tap: 4 global b128 (A) + 8 ds b128 (B)
// feed 8 WMMAs (0.5 LDS pairs per WMMA).
// MODE 0 (conv1): out = bf16(lrelu(acc*isg + bias + ns*noise) * s_next) into
//                 the NHWC interior of out_bf (conv2's input).
// MODE 1 (conv2): out = lrelu(acc*isg + bias + ns*noise) written fp32 NCHW.
// ---------------------------------------------------------------------------
template <int MODE>
__global__ __launch_bounds__(256) void conv3x3_wmma_kernel(
    const __bf16* __restrict__ in, const __bf16* __restrict__ wp,
    const float* __restrict__ isg, const float* __restrict__ bias,
    const float* __restrict__ noise, const float* __restrict__ nsp,
    const float* __restrict__ s_next,
    __bf16* __restrict__ out_bf, float* __restrict__ out_f)
{
    __shared__ __align__(64) __bf16 tile[2][TILE_ELEMS];

    const int lane = threadIdx.x & 31;
    const int wave = threadIdx.x >> 5;
    const int co_base = blockIdx.x * 256 + wave * 32;   // 32 couts per wave
    const int oy = blockIdx.y;
    const int n  = blockIdx.z;
    const int lx = lane & 15;
    const int hi = lane >> 4;            // 0: lanes 0-15, 1: lanes 16-31

    // LDS byte offset of the tile (flat shared pointer truncated to aperture offset)
    const uint32_t lds_base = (uint32_t)(uintptr_t)(&tile[0][0]);
    const __bf16* row_src = in + ((size_t)(n * PD + oy) * PD) * CH;

    // issue the async global->LDS copy of one 32-channel chunk into buffer `buf`
    auto stage = [&](int c0, int buf) {
        const __bf16* src = row_src + c0;
        uint32_t ldst = lds_base + (uint32_t)(buf * (TILE_ELEMS * 2));
#pragma unroll
        for (int k = 0; k < 4; ++k) {
            int i = (int)threadIdx.x + k * 256;
            if (i < TILE_B128) {
                int e0 = i * 8;                 // first bf16 element of this b128
                int cc = e0 & 31;               // channel within chunk (0,8,16,24)
                int xy = e0 >> 5;
                int xx = xy % PD;
                int yy = xy / PD;
                uint64_t g = (uint64_t)(uintptr_t)(src + ((size_t)yy * PD + xx) * CH + cc);
                uint32_t l = ldst + (uint32_t)(i * 16);
                asm volatile("global_load_async_to_lds_b128 %0, %1, off"
                             :: "v"(l), "v"(g) : "memory");
            }
        }
    };

    v8f acc[2][4] = {};    // [co sub-tile][position sub-tile]

    // A fragment sources: rows co_base+lx and co_base+16+lx; lane group picks
    // K 0-15 (lanes 0-15) vs 16-31 (lanes 16-31)
    const __bf16* wrow0 = wp + (size_t)(co_base + lx) * (TAPS * CH) + hi * 16;
    const __bf16* wrow1 = wrow0 + (size_t)16 * (TAPS * CH);

    stage(0, 0);

    // preload A for (chunk 0, tap 0)
    v8bf alo0 = *(const v8bf*)(wrow0);
    v8bf ahi0 = *(const v8bf*)(wrow0 + 8);
    v8bf alo1 = *(const v8bf*)(wrow1);
    v8bf ahi1 = *(const v8bf*)(wrow1 + 8);

    for (int ch = 0; ch < CH / 32; ++ch) {
        const int c0 = ch * 32;
        asm volatile("s_wait_asynccnt 0x0" ::: "memory");
        __syncthreads();
        if (ch + 1 < CH / 32) stage(c0 + 32, (ch + 1) & 1);

        const __bf16* lbuf = &tile[ch & 1][0];
#pragma unroll
        for (int tap = 0; tap < TAPS; ++tap) {
            const int ky = tap / 3, kx = tap % 3;

            // materialize current A fragments, then prefetch the next ones
            v16bf a0, a1;
#pragma unroll
            for (int i = 0; i < 8; ++i) {
                a0[i] = alo0[i]; a0[8 + i] = ahi0[i];
                a1[i] = alo1[i]; a1[8 + i] = ahi1[i];
            }
            {
                size_t off;
                if (tap < TAPS - 1)        off = (size_t)(tap + 1) * CH + c0;
                else if (ch + 1 < CH / 32) off = (size_t)(c0 + 32);   // next chunk, tap 0
                else                       off = 0;                   // harmless reload
                alo0 = *(const v8bf*)(wrow0 + off);
                ahi0 = *(const v8bf*)(wrow0 + off + 8);
                alo1 = *(const v8bf*)(wrow1 + off);
                ahi1 = *(const v8bf*)(wrow1 + off + 8);
            }

            // B fragments from LDS: lanes 0-15 take K 0-15, lanes 16-31 take K 16-31
            const __bf16* pB = lbuf + ((size_t)(ky * PD + kx) * 32 + hi * 16);
            v16bf b0 = *(const v16bf*)(pB + (size_t)(lx) * 32);
            v16bf b1 = *(const v16bf*)(pB + (size_t)(lx + 16) * 32);
            v16bf b2 = *(const v16bf*)(pB + (size_t)(lx + 32) * 32);
            v16bf b3 = *(const v16bf*)(pB + (size_t)(lx + 48) * 32);

            acc[0][0] = WMMA_BF16(a0, b0, acc[0][0]);
            acc[1][0] = WMMA_BF16(a1, b0, acc[1][0]);
            acc[0][1] = WMMA_BF16(a0, b1, acc[0][1]);
            acc[1][1] = WMMA_BF16(a1, b1, acc[1][1]);
            acc[0][2] = WMMA_BF16(a0, b2, acc[0][2]);
            acc[1][2] = WMMA_BF16(a1, b2, acc[1][2]);
            acc[0][3] = WMMA_BF16(a0, b3, acc[0][3]);
            acc[1][3] = WMMA_BF16(a1, b3, acc[1][3]);
        }
    }

    // C/D layout: lane 0-15 -> N=lane, rows M = r (+8 for lanes 16-31)
    const float nsv = nsp[0];
#pragma unroll
    for (int a = 0; a < 2; ++a) {
        const int co_r = co_base + a * 16 + hi * 8;
        float ig[8], bs[8], sn[8];
#pragma unroll
        for (int r = 0; r < 8; ++r) {
            ig[r] = isg[n * CH + co_r + r];
            bs[r] = bias[co_r + r];
            if (MODE == 0) sn[r] = s_next[n * CH + co_r + r];
        }
#pragma unroll
        for (int px = 0; px < 4; ++px) {
            int ox = px * 16 + lx;
            float nz = nsv * noise[(n * HOUT + oy) * HOUT + ox];
            if (MODE == 0) {
                v8bf ov;
#pragma unroll
                for (int r = 0; r < 8; ++r) {
                    float v = lrelu(acc[a][px][r] * ig[r] + bs[r] + nz);
                    ov[r] = (__bf16)(v * sn[r]);
                }
                __bf16* po = out_bf + (((size_t)(n * PD + oy + 1) * PD) + ox + 1) * CH + co_r;
                *(v8bf*)po = ov;
            } else {
#pragma unroll
                for (int r = 0; r < 8; ++r) {
                    float v = lrelu(acc[a][px][r] * ig[r] + bs[r] + nz);
                    out_f[(((size_t)n * CH + co_r + r) * HOUT + oy) * HOUT + ox] = v;
                }
            }
        }
    }
}

// ---------------------------------------------------------------------------
// toRGB (1x1 conv, no demod) + LeakyReLU + bilinear-upsampled y
// ---------------------------------------------------------------------------
__global__ void rgb_kernel(const float* __restrict__ h, const float* __restrict__ y,
                           const float* __restrict__ rgb_w, const float* __restrict__ rgb_b,
                           const float* __restrict__ s_rgb, float* __restrict__ out2)
{
    int t = blockIdx.x * blockDim.x + threadIdx.x;
    if (t >= NB * 3 * HOUT * HOUT) return;
    int p = t & (HOUT * HOUT - 1);
    int r = t >> 12;
    int c = r % 3;
    int n = r / 3;
    int ox = p & 63, oy = p >> 6;

    float acc = rgb_b[c];
    const float* hp = h + (size_t)n * CH * (HOUT * HOUT) + p;
    const float* wc = rgb_w + c * CH;
    const float* sp = s_rgb + n * CH;
    for (int co = 0; co < CH; ++co)
        acc += wc[co] * sp[co] * hp[(size_t)co * (HOUT * HOUT)];
    acc = lrelu(acc);

    // bilinear 32->64, align_corners=False (edge clamp)
    float sy = oy * 0.5f - 0.25f, sx = ox * 0.5f - 0.25f;
    int y0 = (int)floorf(sy); float fy = sy - (float)y0;
    int x0 = (int)floorf(sx); float fx = sx - (float)x0;
    int y1 = y0 + 1, x1 = x0 + 1;
    y0 = y0 < 0 ? 0 : (y0 > 31 ? 31 : y0);
    y1 = y1 < 0 ? 0 : (y1 > 31 ? 31 : y1);
    x0 = x0 < 0 ? 0 : (x0 > 31 ? 31 : x0);
    x1 = x1 < 0 ? 0 : (x1 > 31 ? 31 : x1);
    const float* yp = y + ((size_t)n * 3 + c) * (HIN * HIN);
    float v = (1.f - fy) * ((1.f - fx) * yp[y0 * 32 + x0] + fx * yp[y0 * 32 + x1]) +
              fy        * ((1.f - fx) * yp[y1 * 32 + x0] + fx * yp[y1 * 32 + x1]);
    out2[t] = v + acc;
}

// ---------------------------------------------------------------------------
extern "C" void kernel_launch(void* const* d_in, const int* in_sizes, int n_in,
                              void* d_out, int out_size, void* d_ws, size_t ws_size,
                              hipStream_t stream)
{
    const float* x      = (const float*)d_in[0];
    const float* v      = (const float*)d_in[1];
    const float* y      = (const float*)d_in[2];
    const float* noise1 = (const float*)d_in[3];
    const float* noise2 = (const float*)d_in[4];
    const float* up_w   = (const float*)d_in[5];
    const float* up_b   = (const float*)d_in[6];
    const float* up_sw  = (const float*)d_in[7];
    const float* up_sb  = (const float*)d_in[8];
    const float* c_w    = (const float*)d_in[9];
    const float* c_b    = (const float*)d_in[10];
    const float* c_sw   = (const float*)d_in[11];
    const float* c_sb   = (const float*)d_in[12];
    const float* rgb_w  = (const float*)d_in[13];
    const float* rgb_b  = (const float*)d_in[14];
    const float* rgb_sw = (const float*)d_in[15];
    const float* rgb_sb = (const float*)d_in[16];
    const float* nstr   = (const float*)d_in[17];

    float* h_out   = (float*)d_out;                       // [8][512][64][64]
    float* rgb_out = (float*)d_out + (size_t)NB * CH * HOUT * HOUT;

    // workspace carve (~83 MB total)
    char* w = (char*)d_ws;
    auto carve = [&](size_t bytes) {
        void* p = (void*)w;
        w += (bytes + 255) & ~(size_t)255;
        return p;
    };
    float*  s_up   = (float*)carve(NB * CH * 4);
    float*  s_c    = (float*)carve(NB * CH * 4);
    float*  s_rgb  = (float*)carve(NB * CH * 4);
    float*  isg_up = (float*)carve(NB * CH * 4);
    float*  isg_c  = (float*)carve(NB * CH * 4);
    float*  wsq_up = (float*)carve((size_t)CH * CH * 4);
    float*  wsq_c  = (float*)carve((size_t)CH * CH * 4);
    __bf16* wp_up  = (__bf16*)carve((size_t)CH * TAPS * CH * 2);
    __bf16* wp_c   = (__bf16*)carve((size_t)CH * TAPS * CH * 2);
    const size_t ABYTES = (size_t)NB * PD * PD * CH * 2;  // 35.7 MB each
    __bf16* a1     = (__bf16*)carve(ABYTES);
    __bf16* a2     = (__bf16*)carve(ABYTES);

    // zero padded NHWC buffers (halo + zero-stuffing baseline)
    hipMemsetAsync(a1, 0, ABYTES, stream);
    hipMemsetAsync(a2, 0, ABYTES, stream);

    {   // style vectors
        int tot = 3 * NB * CH;
        modstyle_kernel<<<(tot + 255) / 256, 256, 0, stream>>>(
            v, up_sw, up_sb, c_sw, c_sb, rgb_sw, rgb_sb, s_up, s_c, s_rgb);
    }
    {   // per-(co,ci) squared weight sums
        int tot = 2 * CH * CH;
        wsq_kernel<<<(tot + 255) / 256, 256, 0, stream>>>(up_w, c_w, wsq_up, wsq_c);
    }
    {   // bf16 weight packing
        int tot = 2 * CH * TAPS * CH;
        wpack_kernel<<<(tot + 255) / 256, 256, 0, stream>>>(up_w, c_w, wp_up, wp_c);
    }
    {   // demod reciprocals
        int tot = 2 * NB * CH;
        invsigma_kernel<<<(tot + 255) / 256, 256, 0, stream>>>(
            s_up, s_c, wsq_up, wsq_c, isg_up, isg_c);
    }
    {   // zero-stuffed modulated input for the transposed conv
        int tot = NB * HIN * HIN * CH;
        stuff_kernel<<<(tot + 255) / 256, 256, 0, stream>>>(x, s_up, a1);
    }

    dim3 cgrid(CH / 256, HOUT, NB);
    // conv1 (transposed conv as dense conv over stuffed input) -> a2 (conv2 input)
    conv3x3_wmma_kernel<0><<<cgrid, 256, 0, stream>>>(
        a1, wp_up, isg_up, up_b, noise1, nstr, s_c, a2, nullptr);
    // conv2 -> h (fp32, first output)
    conv3x3_wmma_kernel<1><<<cgrid, 256, 0, stream>>>(
        a2, wp_c, isg_c, c_b, noise2, nstr, nullptr, nullptr, h_out);

    {   // toRGB + bilinear(y) -> second output
        int tot = NB * 3 * HOUT * HOUT;
        rgb_kernel<<<(tot + 255) / 256, 256, 0, stream>>>(
            h_out, y, rgb_w, rgb_b, s_rgb, rgb_out);
    }
}
